// HybridMsgPass_65833258713547
// MI455X (gfx1250) — matrix operand order repaired and checked by reading, not compiled, not run on hardware
//
#include <hip/hip_runtime.h>

typedef __bf16 bf16_t;
typedef __attribute__((ext_vector_type(16))) bf16_t v16bf;
typedef __attribute__((ext_vector_type(8)))  bf16_t bf8;
typedef __attribute__((ext_vector_type(4)))  bf16_t bf4;
typedef __attribute__((ext_vector_type(8)))  float  v8f;
typedef __attribute__((ext_vector_type(4)))  unsigned int u32x4;
typedef __attribute__((ext_vector_type(8)))  int i32x8;
typedef __attribute__((ext_vector_type(4)))  int i32x4;

constexpr int kH       = 256;   // hidden size
constexpr int kMB      = 64;    // rows per block
constexpr int kThreads = 128;   // 4 waves (wave32)

// LDS layout in bf16 elements (rows padded to dodge bank conflicts)
// feat[64][1032] holds  [ h | h' | h-h' | h*h' ]  during phase 0/1, then the
// last two 256-wide regions are reused as [ fuse | bf16(f) ] for phase 2
// (same-wave overwrite only -> ordered by DScnt, no barrier needed).
constexpr int FEAT_STRIDE = 1032;                    // 4H + 8
constexpr int B_STRIDE    = 40;                      // 32 elems + 8 pad (64B+16B)
constexpr int FUSE_COL    = 512;                     // fuse aliases (h-h')
constexpr int F_COL       = 768;                     // bf16(f) aliases (h*h')
constexpr int BST_OFF     = kMB * FEAT_STRIDE;       // 66048 elems
constexpr int BST_ELEMS   = 6 * 64 * B_STRIDE;       // 15360 elems (phase-2 max)
constexpr int LDS_ELEMS   = BST_OFF + BST_ELEMS;     // 81408
constexpr int LDS_BYTES   = LDS_ELEMS * 2;           // 162816 B -> 2 blocks/WGP

// sched_group_barrier instruction-class masks
#define SG_MFMA    0x008   // MFMA / WMMA
#define SG_DSREAD  0x100   // DS read

#if defined(__has_builtin)
#if __has_builtin(__builtin_amdgcn_sched_group_barrier)
#define SGB(mask, n) __builtin_amdgcn_sched_group_barrier((mask), (n), 0)
#endif
#endif
#ifndef SGB
#define SGB(mask, n)
#endif

__device__ __forceinline__ float sig_(float x) { return 1.0f / (1.0f + __expf(-x)); }
__device__ __forceinline__ float tanh_(float x) {
  float e = __expf(-2.0f * x);
  return (1.0f - e) / (1.0f + e);
}

__device__ __forceinline__ v16bf load_frag(const bf16_t* p) {
  bf8 a = *(const bf8*)p;        // 16B
  bf8 b = *(const bf8*)(p + 8);  // next 16B (frag is 32 contiguous bytes)
  return __builtin_shufflevector(a, b, 0, 1, 2, 3, 4, 5, 6, 7,
                                       8, 9, 10, 11, 12, 13, 14, 15);
}

__device__ __forceinline__ v16bf load_frag_split(const bf16_t* p) {
  // A-fragment: halves at +0 and +16 elements
  bf8 a = *(const bf8*)p;
  bf8 b = *(const bf8*)(p + 16);
  return __builtin_shufflevector(a, b, 0, 1, 2, 3, 4, 5, 6, 7,
                                       8, 9, 10, 11, 12, 13, 14, 15);
}

__device__ __forceinline__ v8f wmma_bf16(v16bf a, v16bf b, v8f c) {
  return __builtin_amdgcn_wmma_f32_16x16x32_bf16(false, a, false, b, (short)0,
                                                 c, false, false);
}

// ---- Tensor Data Mover: 2D bf16 tile (rows x kElems) global -> LDS with
// per-row padding (64B data + 16B pad == B_STRIDE of 40 elems).  D# per
// CDNA5 ISA ss8.3/8.4; groups 2/3 zero (2-D tile).
// clang-23 toolchain: 6-arg builtin (g0, g1, g2, g3, g4, cpol).
__device__ __forceinline__ void tdm_load_2d(const bf16_t* gsrc,
                                            unsigned int lds_byte_off,
                                            int kElems, int rows,
                                            int rowStrideElems) {
  unsigned long long ga = (unsigned long long)gsrc;
  u32x4 g0;
  g0[0] = 1u;                                              // count=1 (valid)
  g0[1] = lds_byte_off;                                    // lds_addr (bytes)
  g0[2] = (unsigned int)ga;                                // global_addr[31:0]
  g0[3] = (unsigned int)((ga >> 32) & 0x01FFFFFFu) | (2u << 30);  // +type=2
  i32x8 g1;
  g1[0] = (1 << 16)      // data_size = 2 bytes
        | (1 << 20)      // pad_enable
        | (3 << 22)      // pad_interval: 16 dwords (64B) between pads
        | (3 << 25);     // pad_amount: 4 dwords (16B)
  g1[1] = (kElems & 0xFFFF) << 16;                         // tensor_dim0 lo
  g1[2] = ((kElems >> 16) & 0xFFFF) | ((rows & 0xFFFF) << 16);  // d0 hi, d1 lo
  g1[3] = ((rows >> 16) & 0xFFFF) | ((kElems & 0xFFFF) << 16);  // d1 hi, tile_dim0
  g1[4] = rows & 0xFFFF;                                   // tile_dim1 (dim2=0)
  g1[5] = rowStrideElems;                                  // tensor_dim0_stride
  g1[6] = 0;
  g1[7] = 0;
  i32x4 z4 = {0, 0, 0, 0};
  i32x8 z8 = {0, 0, 0, 0, 0, 0, 0, 0};
  __builtin_amdgcn_tensor_load_to_lds(g0, g1, z4, z4, z8, 0);
}

// ---------------- weight fp32 -> bf16 conversion (weights stay L2-resident) --
__global__ void cvt_weights(const float* __restrict__ Wz,
                            const float* __restrict__ Wih,
                            const float* __restrict__ Whh,
                            bf16_t* __restrict__ ws) {
  const int nz = 256 * 1024, nih = 768 * 256;
  int i = blockIdx.x * 256 + threadIdx.x;
  if (i < nz)                ws[i] = (bf16_t)Wz[i];
  else if (i < nz + nih)     ws[i] = (bf16_t)Wih[i - nz];
  else if (i < nz + 2 * nih) ws[i] = (bf16_t)Whh[i - nz - nih];
}

// ---------------- fused gated-fuse + GRU cell ---------------------------------
__global__ __launch_bounds__(kThreads, 1)
void fused_gru(const float* __restrict__ f,   const float* __restrict__ hv,
               const float* __restrict__ hvp, const float* __restrict__ Wz_b,
               const float* __restrict__ b_ih, const float* __restrict__ b_hh,
               const bf16_t* __restrict__ Wzb, const bf16_t* __restrict__ Wihb,
               const bf16_t* __restrict__ Whhb, float* __restrict__ out) {
  extern __shared__ bf16_t smem[];
  bf16_t* feat = smem;              // [64][1032]
  bf16_t* Bst  = smem + BST_OFF;    // weight chunk staging (TDM destination)

  const int tid     = threadIdx.x;
  const int lane    = tid & 31;
  const int wv      = tid >> 5;       // wave id 0..3, owns rows wv*16..+15
  const int lh      = lane & 15;
  const int half    = lane >> 4;      // 0 or 1
  const int rowBase = blockIdx.x * kMB;

  // ---- Phase 0: build bf16 feature matrix in LDS ----
  for (int p = tid; p < kMB * (kH / 4); p += kThreads) {
    int r  = p >> 6;
    int c4 = (p & 63) << 2;
    const float4 h4  = *(const float4*)(hv  + (size_t)(rowBase + r) * kH + c4);
    const float4 hp4 = *(const float4*)(hvp + (size_t)(rowBase + r) * kH + c4);
    bf16_t* fr = feat + r * FEAT_STRIDE;
    bf4 vh, vp, vd, vm;
    vh.x = (bf16_t)h4.x;  vh.y = (bf16_t)h4.y;  vh.z = (bf16_t)h4.z;  vh.w = (bf16_t)h4.w;
    vp.x = (bf16_t)hp4.x; vp.y = (bf16_t)hp4.y; vp.z = (bf16_t)hp4.z; vp.w = (bf16_t)hp4.w;
    vd.x = (bf16_t)(h4.x - hp4.x); vd.y = (bf16_t)(h4.y - hp4.y);
    vd.z = (bf16_t)(h4.z - hp4.z); vd.w = (bf16_t)(h4.w - hp4.w);
    vm.x = (bf16_t)(h4.x * hp4.x); vm.y = (bf16_t)(h4.y * hp4.y);
    vm.z = (bf16_t)(h4.z * hp4.z); vm.w = (bf16_t)(h4.w * hp4.w);
    *(bf4*)(fr + c4)       = vh;
    *(bf4*)(fr + 256 + c4) = vp;
    *(bf4*)(fr + 512 + c4) = vd;
    *(bf4*)(fr + 768 + c4) = vm;
  }

  // ---- Phase 1: z = sigmoid(z_ @ Wz^T + b); fuse = z*h + (1-z)*h' ----
  v8f acc[16];
  const v8f zero8 = {0.f, 0.f, 0.f, 0.f, 0.f, 0.f, 0.f, 0.f};
#pragma unroll
  for (int t = 0; t < 16; ++t) acc[t] = zero8;

  for (int kc = 0; kc < 32; ++kc) {
    // A-frag preload: feat rows are stable, overlaps TDM + barrier
    const bf16_t* ar = feat + (wv * 16 + lh) * FEAT_STRIDE + kc * 32 + half * 8;
    v16bf av = load_frag_split(ar);
    __syncthreads();          // previous chunk fully consumed
    if (wv == 0) {
      // one TDM descriptor moves the whole 256x32 Wz tile into padded LDS
      tdm_load_2d(Wzb + kc * 32, (unsigned int)(BST_OFF * 2), 32, 256, 1024);
      __builtin_amdgcn_s_wait_tensorcnt(0);
    }
    __syncthreads();          // tile visible to all waves
    const bf16_t* bb = Bst + lh * B_STRIDE + half * 8;
#pragma unroll
    for (int t = 0; t < 16; ++t) {
      v16bf bv = load_frag_split(bb + t * 16 * B_STRIDE);
      acc[t] = wmma_bf16(av, bv, acc[t]);
    }
    // pin a software pipeline: 4 fragments (8 ds reads) of lookahead
    SGB(SG_DSREAD, 8);
#pragma unroll
    for (int i = 0; i < 12; ++i) {
      SGB(SG_MFMA, 1);
      SGB(SG_DSREAD, 2);
    }
#pragma unroll
    for (int i = 0; i < 4; ++i) SGB(SG_MFMA, 1);
  }

  // fuse = z*h + (1-z)*h' -> overwrite (h-h') region (own rows only)
#pragma unroll
  for (int t = 0; t < 16; ++t) {
    int coln = t * 16 + lh;
    float bz = Wz_b[coln];
#pragma unroll
    for (int v = 0; v < 8; ++v) {
      int m = v + half * 8;
      float z  = sig_(acc[t][v] + bz);
      float h  = (float)feat[(wv * 16 + m) * FEAT_STRIDE + coln];
      float hp = (float)feat[(wv * 16 + m) * FEAT_STRIDE + 256 + coln];
      feat[(wv * 16 + m) * FEAT_STRIDE + FUSE_COL + coln] =
          (bf16_t)(z * h + (1.0f - z) * hp);
    }
  }

  // stage bf16(f) for own rows -> overwrite (h*h') region (coalesced)
  for (int j = lane; j < 16 * (kH / 4); j += 32) {
    int r  = j >> 6;
    int c4 = (j & 63) << 2;
    const float4 f4 =
        *(const float4*)(f + (size_t)(rowBase + wv * 16 + r) * kH + c4);
    bf4 vf;
    vf.x = (bf16_t)f4.x; vf.y = (bf16_t)f4.y;
    vf.z = (bf16_t)f4.z; vf.w = (bf16_t)f4.w;
    *(bf4*)(feat + (wv * 16 + r) * FEAT_STRIDE + F_COL + c4) = vf;
  }

  // ---- Phase 2: gi = f@W_ih^T, gh = fuse@W_hh^T, GRU gates, output ----
  for (int c0 = 0; c0 < kH; c0 += 64) {
    v8f gacc[6][4];   // [ih_r, ih_z, ih_n, hh_r, hh_z, hh_n][tile]
#pragma unroll
    for (int g = 0; g < 6; ++g)
#pragma unroll
      for (int t = 0; t < 4; ++t) gacc[g][t] = zero8;

    for (int kc = 0; kc < 8; ++kc) {
      // A-frag preloads (stable LDS regions), each reused by 12 WMMAs
      const bf16_t* frp =
          feat + (wv * 16 + lh) * FEAT_STRIDE + F_COL + kc * 32 + half * 8;
      const bf16_t* urp =
          feat + (wv * 16 + lh) * FEAT_STRIDE + FUSE_COL + kc * 32 + half * 8;
      v16bf af = load_frag_split(frp);
      v16bf au = load_frag_split(urp);
      __syncthreads();
      // six 64x32 tiles (ih_r, ih_z, ih_n, hh_r, hh_z, hh_n): two TDM
      // descriptors issued by each of waves 0..2
      if (wv < 3) {
#pragma unroll
        for (int s = 0; s < 2; ++s) {
          int g = wv * 2 + s;
          const bf16_t* base = (g < 3) ? Wihb : Whhb;
          const bf16_t* gsrc =
              base + (size_t)((g % 3) * 256 + c0) * kH + kc * 32;
          tdm_load_2d(gsrc,
                      (unsigned int)((BST_OFF + g * 64 * B_STRIDE) * 2),
                      32, 64, 256);
        }
        __builtin_amdgcn_s_wait_tensorcnt(0);
      }
      __syncthreads();
      const bf16_t* bb = Bst + lh * B_STRIDE + half * 8;
#pragma unroll
      for (int g = 0; g < 6; ++g) {
        const v16bf A = (g < 3) ? af : au;
        const bf16_t* gb = bb + g * 64 * B_STRIDE;
#pragma unroll
        for (int t = 0; t < 4; ++t) {
          v16bf bv = load_frag_split(gb + t * 16 * B_STRIDE);
          gacc[g][t] = wmma_bf16(A, bv, gacc[g][t]);
        }
      }
      // pin pipeline: 4-fragment lookahead over 24 WMMAs / 48 DS reads
      SGB(SG_DSREAD, 8);
#pragma unroll
      for (int i = 0; i < 20; ++i) {
        SGB(SG_MFMA, 1);
        SGB(SG_DSREAD, 2);
      }
#pragma unroll
      for (int i = 0; i < 4; ++i) SGB(SG_MFMA, 1);
    }
    // gate math + output
#pragma unroll
    for (int t = 0; t < 4; ++t) {
      int coln = c0 + t * 16 + lh;
      float bir = b_ih[coln], biz = b_ih[256 + coln], bin = b_ih[512 + coln];
      float bhr = b_hh[coln], bhz = b_hh[256 + coln], bhn = b_hh[512 + coln];
#pragma unroll
      for (int v = 0; v < 8; ++v) {
        int m = v + half * 8;
        float r  = sig_(gacc[0][t][v] + bir + gacc[3][t][v] + bhr);
        float u  = sig_(gacc[1][t][v] + biz + gacc[4][t][v] + bhz);
        float nn = tanh_(gacc[2][t][v] + bin + r * (gacc[5][t][v] + bhn));
        float fu =
            (float)feat[(wv * 16 + m) * FEAT_STRIDE + FUSE_COL + coln];
        out[(size_t)(rowBase + wv * 16 + m) * kH + coln] =
            (1.0f - u) * nn + u * fu;
      }
    }
  }
}

extern "C" void kernel_launch(void* const* d_in, const int* in_sizes, int n_in,
                              void* d_out, int out_size, void* d_ws, size_t ws_size,
                              hipStream_t stream) {
  const float* f   = (const float*)d_in[0];
  const float* hv  = (const float*)d_in[1];
  const float* hvp = (const float*)d_in[2];
  const float* Wz  = (const float*)d_in[3];
  const float* Wzb = (const float*)d_in[4];
  const float* Wih = (const float*)d_in[5];
  const float* Whh = (const float*)d_in[6];
  const float* bih = (const float*)d_in[7];
  const float* bhh = (const float*)d_in[8];
  float* out = (float*)d_out;

  bf16_t* wb     = (bf16_t*)d_ws;
  bf16_t* Wzb16  = wb;
  bf16_t* Wihb16 = wb + 256 * 1024;
  bf16_t* Whhb16 = wb + 256 * 1024 + 768 * 256;

  const int totalW = 256 * 1024 + 2 * 768 * 256;  // 655360
  cvt_weights<<<(totalW + 255) / 256, 256, 0, stream>>>(Wz, Wih, Whh, wb);

  const int N = in_sizes[0] / kH;        // 200000
  const int nBlocks = N / kMB;           // 3125 (exact)
  fused_gru<<<nBlocks, kThreads, LDS_BYTES, stream>>>(
      f, hv, hvp, Wzb, bih, bhh, Wzb16, Wihb16, Whhb16, out);
}
